// AttnBlock_22428319220089
// MI455X (gfx1250) — matrix operand order repaired and checked by reading, not compiled, not run on hardware
//
#include <hip/hip_runtime.h>
#include <math.h>

// ---------------------------------------------------------------------------
// Types for CDNA5 WMMA (wave32, 16x16x32 bf16 -> f32)
// ---------------------------------------------------------------------------
typedef __bf16 bf16_t;
typedef __attribute__((ext_vector_type(16))) __bf16 v16bf;
typedef __attribute__((ext_vector_type(8)))  __bf16 v8bf;
typedef __attribute__((ext_vector_type(8)))  float  v8f;

#define C_DIM  512
#define N_TOK  4096
#define BATCH  4
#define M_TOT  (BATCH * N_TOK)   // 16384
#define NGROUP 32
#define CGRP   16                // C / GROUPS

__device__ __forceinline__ v8f wmma_bf16(v16bf a, v16bf b, v8f c) {
  // emits v_wmma_f32_16x16x32_bf16
  return __builtin_amdgcn_wmma_f32_16x16x32_bf16(false, a, false, b, (short)0, c,
                                                 false, false);
}

// A-fragment (16x32 bf16) from row-major source, leading dim ld.
// lane<16: elems 0..7 = K 0..7, elems 8..15 = K 16..23 (row = lane)
// lane>=16: elems 0..7 = K 8..15, elems 8..15 = K 24..31 (row = lane-16)
// Two contiguous 16B loads per lane.
__device__ __forceinline__ v16bf load_a_frag(const bf16_t* __restrict__ base, int ld) {
  const int lane = threadIdx.x & 31;
  const bf16_t* p = base + (lane & 15) * ld + ((lane >> 4) << 3);
  union { v16bf v; v8bf h[2]; } u;
  u.h[0] = *(const v8bf*)p;
  u.h[1] = *(const v8bf*)(p + 16);
  return u.v;
}

// B-fragment (32x16) where the reduction dim is CONTIGUOUS in memory
// (i.e. source is the transpose: rows = output column, cols = K).
// col = lane&15; lanes 0-15 hold K=0..7,16..23, lanes 16-31 hold K=8..15,24..31.
// Two contiguous 16B loads per lane.
__device__ __forceinline__ v16bf load_b_frag_kcontig(const bf16_t* __restrict__ base,
                                                     int ld) {
  const int lane = threadIdx.x & 31;
  const bf16_t* p = base + (lane & 15) * ld + ((lane >> 4) << 3);
  union { v16bf v; v8bf h[2]; } u;
  u.h[0] = *(const v8bf*)p;
  u.h[1] = *(const v8bf*)(p + 16);
  return u.v;
}

// ---------------------------------------------------------------------------
// GroupNorm
// ---------------------------------------------------------------------------
__global__ void gn_stats_kernel(const float* __restrict__ x, float* __restrict__ stats) {
  const int bg = blockIdx.x;          // b*32 + g
  const int b  = bg >> 5;
  const int g  = bg & 31;
  const float* xb = x + (b * N_TOK) * C_DIM + g * CGRP;
  float s = 0.f, ss = 0.f;
  for (int i = threadIdx.x; i < N_TOK * CGRP; i += blockDim.x) {
    float v = xb[(i >> 4) * C_DIM + (i & 15)];
    s += v; ss += v * v;
  }
#pragma unroll
  for (int off = 16; off > 0; off >>= 1) {
    s  += __shfl_xor(s, off, 32);
    ss += __shfl_xor(ss, off, 32);
  }
  __shared__ float red[2][8];
  const int wave = threadIdx.x >> 5;
  if ((threadIdx.x & 31) == 0) { red[0][wave] = s; red[1][wave] = ss; }
  __syncthreads();
  if (threadIdx.x == 0) {
    float S = 0.f, SS = 0.f;
#pragma unroll
    for (int w = 0; w < 8; ++w) { S += red[0][w]; SS += red[1][w]; }
    const float n = (float)(N_TOK * CGRP);
    float mean = S / n;
    float var  = SS / n - mean * mean;
    stats[bg * 2 + 0] = mean;
    stats[bg * 2 + 1] = rsqrtf(var + 1e-6f);
  }
}

__global__ void gn_apply_kernel(const float* __restrict__ x,
                                const float* __restrict__ stats,
                                const float* __restrict__ gscale,
                                const float* __restrict__ gbias,
                                bf16_t* __restrict__ h, int total) {
  for (int i = blockIdx.x * blockDim.x + threadIdx.x; i < total;
       i += gridDim.x * blockDim.x) {
    int c = i & (C_DIM - 1);
    int b = i >> 21;                 // / (N_TOK * C_DIM) = / 2^21
    int g = c >> 4;
    float mean = stats[(b * NGROUP + g) * 2 + 0];
    float rstd = stats[(b * NGROUP + g) * 2 + 1];
    float v = (x[i] - mean) * rstd * gscale[c] + gbias[c];
    h[i] = (bf16_t)v;
  }
}

// Transpose one 512x512 f32 matrix into bf16 (LDS-tiled, coalesced both sides).
__global__ void transpose_to_bf16_kernel(const float* __restrict__ a,
                                         bf16_t* __restrict__ o) {
  __shared__ float tile[32][33];
  const int bx = blockIdx.x * 32, by = blockIdx.y * 32;
  const int tx = threadIdx.x & 31, ty = threadIdx.x >> 5;   // 32 x 8
#pragma unroll
  for (int r = ty; r < 32; r += 8)
    tile[r][tx] = a[(by + r) * C_DIM + bx + tx];
  __syncthreads();
#pragma unroll
  for (int r = ty; r < 32; r += 8)
    o[(bx + r) * C_DIM + by + tx] = (bf16_t)tile[tx][r];
}

// ---------------------------------------------------------------------------
// GEMM: (M x 512 bf16, row-major) @ W, with W supplied TRANSPOSED (N x K
// row-major) so B-fragments are contiguous. Per wave: 16 rows x 64 cols.
// Depth-1 software pipeline: fragments for kc+1 are in flight while the
// WMMAs for kc execute (partial s_wait_loadcnt instead of drain-to-zero).
//   mode 0: outb[row*512+col]      = (acc + bias) * scale   (bf16, row-major)
//   mode 1: outf[row*512+col]      = resid + acc + bias      (f32, row-major)
//   mode 2: outb[col*M_TOT + row]  = acc + bias              (bf16, transposed,
//                                    contiguous v8bf store per lane)
// ---------------------------------------------------------------------------
__global__ void __launch_bounds__(256, 1)
gemm_kernel(const bf16_t* __restrict__ A,
            const bf16_t* __restrict__ Wt,       // 512 x 512, W^T row-major
            const float* __restrict__ bias,
            bf16_t* __restrict__ outb,
            float* __restrict__ outf,
            const float* __restrict__ resid,
            float scale, int mode) {
  const int wave = threadIdx.x >> 5;
  const int lane = threadIdx.x & 31;
  const int row0 = (blockIdx.x * 8 + wave) * 16;
  const int col0 = blockIdx.y * 64;
  const v8f vzero = {0.f, 0.f, 0.f, 0.f, 0.f, 0.f, 0.f, 0.f};
  v8f acc[4];
#pragma unroll
  for (int t = 0; t < 4; ++t) acc[t] = vzero;

  const bf16_t* arow = A + row0 * C_DIM;
  const bf16_t* wcol = Wt + col0 * C_DIM;

  v16bf a_cur = load_a_frag(arow, C_DIM);
  v16bf b_cur[4];
#pragma unroll
  for (int t = 0; t < 4; ++t)
    b_cur[t] = load_b_frag_kcontig(wcol + t * 16 * C_DIM, C_DIM);

#pragma unroll 4
  for (int kc = 0; kc < 16; ++kc) {
    v16bf a_nxt = a_cur;
    v16bf b_nxt[4];
#pragma unroll
    for (int t = 0; t < 4; ++t) b_nxt[t] = b_cur[t];
    if (kc < 15) {
      __builtin_prefetch(arow + (kc + 2) * 32, 0, 3);   // global_prefetch_b8
      a_nxt = load_a_frag(arow + (kc + 1) * 32, C_DIM);
#pragma unroll
      for (int t = 0; t < 4; ++t)
        b_nxt[t] = load_b_frag_kcontig(wcol + t * 16 * C_DIM + (kc + 1) * 32, C_DIM);
    }
#pragma unroll
    for (int t = 0; t < 4; ++t) acc[t] = wmma_bf16(a_cur, b_cur[t], acc[t]);
    a_cur = a_nxt;
#pragma unroll
    for (int t = 0; t < 4; ++t) b_cur[t] = b_nxt[t];
  }

  const int rbase = (lane >> 4) << 3;   // D layout: row = r + 8*(lane/16)
  const int cl    = lane & 15;
#pragma unroll
  for (int t = 0; t < 4; ++t) {
    const int col = col0 + t * 16 + cl;
    const float bv = bias[col];
    if (mode == 2) {
      v8bf pack;
#pragma unroll
      for (int r = 0; r < 8; ++r) pack[r] = (bf16_t)(acc[t][r] + bv);
      *(v8bf*)(outb + (size_t)col * M_TOT + row0 + rbase) = pack;
    } else if (mode == 0) {
#pragma unroll
      for (int r = 0; r < 8; ++r)
        outb[(row0 + rbase + r) * C_DIM + col] = (bf16_t)((acc[t][r] + bv) * scale);
    } else {
#pragma unroll
      for (int r = 0; r < 8; ++r) {
        const int idx = (row0 + rbase + r) * C_DIM + col;
        outf[idx] = resid[idx] + acc[t][r] + bv;
      }
    }
  }
}

// ---------------------------------------------------------------------------
// Attention phase 1: per-row softmax stats (m, l) with online updates.
// One wave handles 16 query rows; all Q A-fragments stay resident (128 VGPRs).
// K B-fragments are double-buffered across the k-chunk loop.
// ---------------------------------------------------------------------------
__global__ void __launch_bounds__(256, 1)
attn_stats_kernel(const bf16_t* __restrict__ q,
                  const bf16_t* __restrict__ k,
                  float* __restrict__ ml) {
  const int wave = threadIdx.x >> 5;
  const int lane = threadIdx.x & 31;
  const int b    = blockIdx.y;
  const int q0   = (blockIdx.x * 8 + wave) * 16;
  const bf16_t* qb = q + (b * N_TOK + q0) * C_DIM;
  const bf16_t* kb = k + b * N_TOK * C_DIM;

  v16bf qa[16];
#pragma unroll
  for (int kc = 0; kc < 16; ++kc) qa[kc] = load_a_frag(qb + kc * 32, C_DIM);

  float m[8], l[8];
#pragma unroll
  for (int r = 0; r < 8; ++r) { m[r] = -3.0e38f; l[r] = 0.f; }
  const v8f vzero = {0.f, 0.f, 0.f, 0.f, 0.f, 0.f, 0.f, 0.f};

  for (int jt = 0; jt < N_TOK / 16; ++jt) {
    v8f s = vzero;
    const bf16_t* kt = kb + (jt * 16) * C_DIM;   // K row-major: K^T is k-contig
    v16bf bcur = load_b_frag_kcontig(kt, C_DIM);
#pragma unroll
    for (int kc = 0; kc < 16; ++kc) {
      v16bf bnxt = bcur;
      if (kc < 15) bnxt = load_b_frag_kcontig(kt + (kc + 1) * 32, C_DIM);
      s = wmma_bf16(qa[kc], bcur, s);
      bcur = bnxt;
    }
#pragma unroll
    for (int r = 0; r < 8; ++r) {
      float sv = s[r];
      float mn = fmaxf(m[r], sv);
      l[r] = l[r] * __expf(m[r] - mn) + __expf(sv - mn);
      m[r] = mn;
    }
  }
  // merge (m,l) across the 16 lanes of each half-wave (same 8 rows per half)
#pragma unroll
  for (int off = 1; off < 16; off <<= 1) {
#pragma unroll
    for (int r = 0; r < 8; ++r) {
      float mo = __shfl_xor(m[r], off, 32);
      float lo = __shfl_xor(l[r], off, 32);
      float mn = fmaxf(m[r], mo);
      l[r] = l[r] * __expf(m[r] - mn) + lo * __expf(mo - mn);
      m[r] = mn;
    }
  }
  if ((lane & 15) == 0) {
    const int rbase = (lane >> 4) << 3;
#pragma unroll
    for (int r = 0; r < 8; ++r) {
      int qi = b * N_TOK + q0 + rbase + r;
      ml[qi * 2 + 0] = m[r];
      ml[qi * 2 + 1] = l[r];
    }
  }
}

// ---------------------------------------------------------------------------
// Attention phase 2: O = softmax(QK^T) V per (query tile, 128-channel slice).
// S recomputed; P staged through wave-private LDS to reach A-fragment layout.
// V is supplied TRANSPOSED [C][M_TOT] so its B-fragments are contiguous.
// K and V fragment streams are double-buffered.
// ---------------------------------------------------------------------------
__global__ void __launch_bounds__(256, 1)
attn_pv_kernel(const bf16_t* __restrict__ q,
               const bf16_t* __restrict__ k,
               const bf16_t* __restrict__ vt,    // [C_DIM][M_TOT]
               const float* __restrict__ ml,
               bf16_t* __restrict__ o) {
  __shared__ bf16_t pbuf[8][16][40];  // per-wave 16x32 P tile, padded
  const int wave = threadIdx.x >> 5;
  const int lane = threadIdx.x & 31;
  const int b    = blockIdx.z;
  const int c0   = blockIdx.y * 128;
  const int q0   = (blockIdx.x * 8 + wave) * 16;
  const bf16_t* qb  = q + (b * N_TOK + q0) * C_DIM;
  const bf16_t* kbp = k + b * N_TOK * C_DIM;
  const bf16_t* vbp = vt + b * N_TOK + (size_t)c0 * M_TOT;

  v16bf qa[16];
#pragma unroll
  for (int kc = 0; kc < 16; ++kc) qa[kc] = load_a_frag(qb + kc * 32, C_DIM);

  const int rbase = (lane >> 4) << 3;
  const int cl    = lane & 15;
  float mrow[8], ilrow[8];
#pragma unroll
  for (int r = 0; r < 8; ++r) {
    int qi = b * N_TOK + q0 + rbase + r;
    mrow[r]  = ml[qi * 2 + 0];
    ilrow[r] = 1.f / ml[qi * 2 + 1];
  }

  const v8f vzero = {0.f, 0.f, 0.f, 0.f, 0.f, 0.f, 0.f, 0.f};
  v8f acc[8];
#pragma unroll
  for (int t = 0; t < 8; ++t) acc[t] = vzero;

  for (int jt = 0; jt < N_TOK / 32; ++jt) {
    const int j0 = jt * 32;
    v8f s0 = vzero, s1 = vzero;
    const bf16_t* kt0 = kbp + j0 * C_DIM;
    const bf16_t* kt1 = kbp + (j0 + 16) * C_DIM;
    v16bf b0c = load_b_frag_kcontig(kt0, C_DIM);
    v16bf b1c = load_b_frag_kcontig(kt1, C_DIM);
#pragma unroll
    for (int kc = 0; kc < 16; ++kc) {
      v16bf b0n = b0c, b1n = b1c;
      if (kc < 15) {
        b0n = load_b_frag_kcontig(kt0 + (kc + 1) * 32, C_DIM);
        b1n = load_b_frag_kcontig(kt1 + (kc + 1) * 32, C_DIM);
      }
      s0 = wmma_bf16(qa[kc], b0c, s0);
      s1 = wmma_bf16(qa[kc], b1c, s1);
      b0c = b0n; b1c = b1n;
    }
    // P = exp(S - m) / l  -> LDS (D layout out, A layout back in)
#pragma unroll
    for (int r = 0; r < 8; ++r) {
      pbuf[wave][rbase + r][cl]      = (bf16_t)(__expf(s0[r] - mrow[r]) * ilrow[r]);
      pbuf[wave][rbase + r][16 + cl] = (bf16_t)(__expf(s1[r] - mrow[r]) * ilrow[r]);
    }
    // V fragments for this key block can be fetched while P settles in LDS
    v16bf vcur = load_b_frag_kcontig(vbp + j0, M_TOT);
    v16bf pa;
    {
      const int row = lane & 15;
      const int kb8 = (lane >> 4) << 3;
      union { v16bf v; v8bf h[2]; } u;
      u.h[0] = *(const v8bf*)&pbuf[wave][row][kb8];        // ds_load_b128
      u.h[1] = *(const v8bf*)&pbuf[wave][row][kb8 + 16];   // ds_load_b128
      pa = u.v;
    }
#pragma unroll
    for (int ct = 0; ct < 8; ++ct) {
      v16bf vnxt = vcur;
      if (ct < 7)
        vnxt = load_b_frag_kcontig(vbp + (size_t)(ct + 1) * 16 * M_TOT + j0, M_TOT);
      acc[ct] = wmma_bf16(pa, vcur, acc[ct]);
      vcur = vnxt;
    }
  }

#pragma unroll
  for (int ct = 0; ct < 8; ++ct)
#pragma unroll
    for (int r = 0; r < 8; ++r) {
      int idx = (b * N_TOK + q0 + rbase + r) * C_DIM + c0 + ct * 16 + cl;
      o[idx] = (bf16_t)acc[ct][r];
    }
}

// ---------------------------------------------------------------------------
// Launch
// ---------------------------------------------------------------------------
extern "C" void kernel_launch(void* const* d_in, const int* in_sizes, int n_in,
                              void* d_out, int out_size, void* d_ws, size_t ws_size,
                              hipStream_t stream) {
  const float* x  = (const float*)d_in[0];
  const float* gs = (const float*)d_in[1];
  const float* gb = (const float*)d_in[2];
  const float* wq = (const float*)d_in[3];
  const float* bq = (const float*)d_in[4];
  const float* wk = (const float*)d_in[5];
  const float* bk = (const float*)d_in[6];
  const float* wv = (const float*)d_in[7];
  const float* bv = (const float*)d_in[8];
  const float* wp = (const float*)d_in[9];
  const float* bp = (const float*)d_in[10];
  float* out = (float*)d_out;

  char* ws = (char*)d_ws;
  const size_t MAT = (size_t)M_TOT * C_DIM * sizeof(bf16_t);   // 16 MiB
  const size_t WSZ = (size_t)C_DIM * C_DIM;                    // per-weight elems
  bf16_t* h   = (bf16_t*)(ws + 0 * MAT);
  bf16_t* qm  = (bf16_t*)(ws + 1 * MAT);
  bf16_t* km  = (bf16_t*)(ws + 2 * MAT);
  bf16_t* vmt = (bf16_t*)(ws + 3 * MAT);   // V transposed: [C_DIM][M_TOT]
  bf16_t* om  = (bf16_t*)(ws + 4 * MAT);
  bf16_t* wbf = (bf16_t*)(ws + 5 * MAT);   // 4 transposed weight mats
  float* gstats = (float*)(ws + 5 * MAT + 4 * WSZ * sizeof(bf16_t));
  float* ml     = gstats + 2 * BATCH * NGROUP;

  const int total = M_TOT * C_DIM;

  gn_stats_kernel<<<BATCH * NGROUP, 256, 0, stream>>>(x, gstats);
  gn_apply_kernel<<<4096, 256, 0, stream>>>(x, gstats, gs, gb, h, total);

  dim3 tgrid(16, 16);
  transpose_to_bf16_kernel<<<tgrid, 256, 0, stream>>>(wq, wbf + 0 * WSZ);
  transpose_to_bf16_kernel<<<tgrid, 256, 0, stream>>>(wk, wbf + 1 * WSZ);
  transpose_to_bf16_kernel<<<tgrid, 256, 0, stream>>>(wv, wbf + 2 * WSZ);
  transpose_to_bf16_kernel<<<tgrid, 256, 0, stream>>>(wp, wbf + 3 * WSZ);

  dim3 ggrid(M_TOT / 128, C_DIM / 64);
  const float qscale = 0.044194173824159216f;  // 512^-0.5 (applied after bias)
  gemm_kernel<<<ggrid, 256, 0, stream>>>(h, wbf + 0 * WSZ, bq, qm, nullptr, nullptr,
                                         qscale, 0);
  gemm_kernel<<<ggrid, 256, 0, stream>>>(h, wbf + 1 * WSZ, bk, km, nullptr, nullptr,
                                         1.f, 0);
  gemm_kernel<<<ggrid, 256, 0, stream>>>(h, wbf + 2 * WSZ, bv, vmt, nullptr, nullptr,
                                         1.f, 2);

  attn_stats_kernel<<<dim3(N_TOK / 128, BATCH), 256, 0, stream>>>(qm, km, ml);
  attn_pv_kernel<<<dim3(N_TOK / 128, C_DIM / 128, BATCH), 256, 0, stream>>>(qm, km, vmt,
                                                                            ml, om);

  gemm_kernel<<<ggrid, 256, 0, stream>>>(om, wbf + 3 * WSZ, bp, nullptr, out, x,
                                         1.f, 1);
}
